// FusedExpertsNetwork_24412594110829
// MI455X (gfx1250) — matrix operand order repaired and checked by reading, not compiled, not run on hardware
//
#include <hip/hip_runtime.h>

typedef __attribute__((ext_vector_type(16))) __bf16          v16bf;
typedef __attribute__((ext_vector_type(2)))  __bf16          v2bf;
typedef __attribute__((ext_vector_type(16))) unsigned short  v16u;
typedef __attribute__((ext_vector_type(8)))  float           v8f;

#define E_ 8
#define C_ 4096
#define M_ 1024
#define H_ 4096
#define O_ 1024
#define BC 32          // capacity rows per block (2 WMMA row-tiles)
#define KC 32          // H-chunk (K-step of GEMM2, N-chunk of GEMM1)

#if defined(__has_builtin)
#if __has_builtin(__builtin_amdgcn_cvt_pk_bf16_f32)
#define HAVE_PK_BF16 1
#endif
#endif

__device__ __forceinline__ unsigned short f2bf(float f) {
    unsigned int u = __builtin_bit_cast(unsigned int, f);
    u += 0x7FFFu + ((u >> 16) & 1u);          // round-to-nearest-even
    return (unsigned short)(u >> 16);
}

// pack two f32 -> two bf16 in one dword (v_cvt_pk_bf16_f32 when available)
__device__ __forceinline__ unsigned int f2bf2(float lo, float hi) {
#ifdef HAVE_PK_BF16
    return __builtin_bit_cast(unsigned int, __builtin_amdgcn_cvt_pk_bf16_f32(lo, hi));
#else
    return (unsigned int)f2bf(lo) | ((unsigned int)f2bf(hi) << 16);
#endif
}

__device__ __forceinline__ v16bf ldfrag(const unsigned short* p) {
    return __builtin_bit_cast(v16bf, *(const v16u*)p);
}

// Dynamic LDS layout:
//   lds_x   : 64 A-frags (2 row-tiles x 32 k-steps) x 512 bf16 = 64 KB
//   lds_w1  : 64 B-frags (8 waves x 2 nt x 4 kt)    x 512 bf16 = 64 KB
//   lds_w2  : 64 B-frags (8 waves x 8 nt)           x 512 bf16 = 64 KB
//   lds_part: 8 waves x 2 rt x 2 nt x 32 lanes x 8 f32         = 32 KB
//   lds_yf  : 2 A-frags x 512 bf16                             =  2 KB
#define LDS_X_SHORTS   (64 * 512)
#define LDS_W1_SHORTS  (64 * 512)
#define LDS_W2_SHORTS  (64 * 512)
#define LDS_PART_FLTS  (8 * 2 * 2 * 32 * 8)
#define LDS_YF_SHORTS  (2 * 512)

__global__ void __launch_bounds__(256, 1)
moe_ffn_fused_kernel(const float* __restrict__ x,  const float* __restrict__ w1,
                     const float* __restrict__ b1, const float* __restrict__ w2,
                     const float* __restrict__ b2, float* __restrict__ out)
{
    extern __shared__ __attribute__((aligned(32))) unsigned char smem[];
    unsigned short* lds_x    = (unsigned short*)smem;
    unsigned short* lds_w1   = lds_x  + LDS_X_SHORTS;
    unsigned short* lds_w2   = lds_w1 + LDS_W1_SHORTS;
    float*          lds_part = (float*)(lds_w2 + LDS_W2_SHORTS);
    unsigned short* lds_yf   = (unsigned short*)(lds_part + LDS_PART_FLTS);

    const int tid  = threadIdx.x;
    const int w    = tid >> 5;            // wave 0..7
    const int lane = tid & 31;
    const int e    = blockIdx.x >> 7;     // C_/BC = 128 slabs per expert
    const int row0 = (blockIdx.x & 127) * BC;

    const size_t xbase  = ((size_t)e * C_ + row0) * M_;
    const size_t w1base = (size_t)e * M_ * H_;
    const size_t w2base = (size_t)e * H_ * O_;

    // ---- stage x slab [32 x 1024] once as bf16 A-fragments.
    // A 16x32: lane = row + 16*Kgrp; half h: K = h + 16*Kgrp.
    // 4 consecutive K (m%4==0) share one lane octet -> one 8-byte store.
    {
        const float4* xv = (const float4*)(x + xbase);
        #pragma unroll 4
        for (int i = 0; i < 32; ++i) {
            const int f4  = i * 256 + tid;         // 8192 float4, coalesced
            const int row = f4 >> 8, m = (f4 & 255) << 2;
            const float4 v = xv[f4];
            const int rt = row >> 4, rr = row & 15;
            const int ktg = m >> 5, r = m & 31;
            const int l = rr + ((r >> 4) << 4);
            const int h = r & 15;
            uint2 pp;
            pp.x = f2bf2(v.x, v.y);
            pp.y = f2bf2(v.z, v.w);
            *(uint2*)&lds_x[(((rt * 32 + ktg) * 32 + l) << 4) + h] = pp;
        }
    }

    // persistent GEMM2 accumulators: wave owns O cols [w*128, w*128+128), 2 row-tiles
    v8f oacc[16];
    #pragma unroll
    for (int i = 0; i < 16; ++i) oacc[i] = (v8f){0,0,0,0,0,0,0,0};

    for (int hc = 0; hc < H_ / KC; ++hc) {
        __syncthreads();   // staging buffers free of previous iteration's readers

        // ---- stage W1 chunk [1024 x 32] -> bf16 B-frags (B: lane = N + 16*(K>=16), h = K&15)
        {
            #pragma unroll 2
            for (int i = 0; i < 32; ++i) {
                const int f4 = i * 256 + tid;          // 8192 float4
                const int m = f4 >> 3, c4 = (f4 & 7) << 2;
                const float4 v = *(const float4*)(w1 + w1base + (size_t)m * H_
                                                  + (size_t)hc * KC + c4);
                const int r = m & 31, kt = (m >> 5) & 3, ww = m >> 7;
                const int hsh = r & 15;
                const int lg  = (r >> 4) << 4;
                const int nt  = c4 >> 4;               // constant for all 4 (c4 % 4 == 0)
                const int n0  = c4 & 15;
                const unsigned int p01 = f2bf2(v.x, v.y);
                const unsigned int p23 = f2bf2(v.z, v.w);
                const int base = ((((ww * 2 + nt) * 4 + kt) * 32 + (n0 + lg)) << 4) + hsh;
                lds_w1[base     ] = (unsigned short)(p01 & 0xffffu);
                lds_w1[base + 16] = (unsigned short)(p01 >> 16);      // ds_store_b16_d16_hi
                lds_w1[base + 32] = (unsigned short)(p23 & 0xffffu);
                lds_w1[base + 48] = (unsigned short)(p23 >> 16);
            }
            // ---- stage W2 chunk [32 x 1024] (fully contiguous) -> bf16 B-frags
            const float4* w2v = (const float4*)(w2 + w2base + (size_t)hc * KC * O_);
            #pragma unroll 2
            for (int i = 0; i < 32; ++i) {
                const int f4 = i * 256 + tid;
                const int k = f4 >> 8, o4 = (f4 & 255) << 2;
                const float4 v = w2v[f4];
                const int hsh = k & 15;
                const int lg  = (k >> 4) << 4;
                const int ww  = o4 >> 7, cc = o4 & 127;   // constant for all 4
                const int nt  = cc >> 4,  n0 = cc & 15;
                const unsigned int p01 = f2bf2(v.x, v.y);
                const unsigned int p23 = f2bf2(v.z, v.w);
                const int base = (((ww * 8 + nt) * 32 + (n0 + lg)) << 4) + hsh;
                lds_w2[base     ] = (unsigned short)(p01 & 0xffffu);
                lds_w2[base + 16] = (unsigned short)(p01 >> 16);
                lds_w2[base + 32] = (unsigned short)(p23 & 0xffffu);
                lds_w2[base + 48] = (unsigned short)(p23 >> 16);
            }
        }
        __syncthreads();

        // ---- GEMM1 partials: wave reduces K slice [w*128, w*128+128), 2 rt x 2 nt tiles
        v8f acc[2][2];
        acc[0][0] = acc[0][1] = acc[1][0] = acc[1][1] = (v8f){0,0,0,0,0,0,0,0};
        #pragma unroll
        for (int kt = 0; kt < 4; ++kt) {
            const int ktg = w * 4 + kt;
            const v16bf bf0 = ldfrag(&lds_w1[(((w * 2 + 0) * 4 + kt) * 32 + lane) << 4]);
            const v16bf bf1 = ldfrag(&lds_w1[(((w * 2 + 1) * 4 + kt) * 32 + lane) << 4]);
            #pragma unroll
            for (int rt = 0; rt < 2; ++rt) {
                const v16bf a = ldfrag(&lds_x[((rt * 32 + ktg) * 32 + lane) << 4]);
                acc[rt][0] = __builtin_amdgcn_wmma_f32_16x16x32_bf16(false, a, false, bf0,
                                                                     (short)0, acc[rt][0], false, false);
                acc[rt][1] = __builtin_amdgcn_wmma_f32_16x16x32_bf16(false, a, false, bf1,
                                                                     (short)0, acc[rt][1], false, false);
            }
        }
        #pragma unroll
        for (int rt = 0; rt < 2; ++rt)
            #pragma unroll
            for (int nt = 0; nt < 2; ++nt)
                #pragma unroll
                for (int r = 0; r < 8; ++r)
                    lds_part[(((((w * 2 + rt) * 2 + nt) * 32) + lane) << 3) + r] = acc[rt][nt][r];
        __syncthreads();

        // ---- cross-wave reduce (column pairs) + bias + ReLU -> packed bf16 y A-fragments
        #pragma unroll
        for (int t = 0; t < 2; ++t) {
            const int pidx = t * 256 + tid;           // 512 column-pairs of y chunk
            const int row = pidx >> 4;                // 32 rows x 16 pairs
            const int c0  = (pidx & 15) << 1;         // even column, pair (c0, c0+1)
            const int rt = row >> 4, rr = row & 15;
            const int r  = rr & 7;
            const int nt = c0 >> 4;
            const int l0 = (c0 & 15) + ((rr >> 3) << 4);   // C/D: lane = N + 16*(M>=8)
            float s0 = 0.f, s1 = 0.f;
            #pragma unroll
            for (int ww = 0; ww < 8; ++ww) {
                const int pb = ((((ww * 2 + rt) * 2 + nt) * 32) + l0) << 3;
                s0 += lds_part[pb + r];
                s1 += lds_part[pb + 8 + r];           // l0+1 -> +8 floats (2addr-combinable)
            }
            const float2 bb = *(const float2*)&b1[(size_t)e * H_ + hc * KC + c0];
            s0 = fmaxf(s0 + bb.x, 0.f);               // ReLU
            s1 = fmaxf(s1 + bb.y, 0.f);
            const int la = rr + ((c0 >> 4) << 4);
            const int h  = c0 & 15;    // even; pair is (h, h+1)
            *(unsigned int*)&lds_yf[((rt * 32 + la) << 4) + h] = f2bf2(s0, s1);
        }
        __syncthreads();

        // ---- GEMM2 K-step: oacc[rt*8+nt] += y_rt x W2_nt (B-frag reused across rt)
        const v16bf ay0 = ldfrag(&lds_yf[(0 * 32 + lane) << 4]);
        const v16bf ay1 = ldfrag(&lds_yf[(1 * 32 + lane) << 4]);
        #pragma unroll
        for (int nt = 0; nt < 8; ++nt) {
            const v16bf bf = ldfrag(&lds_w2[((w * 8 + nt) * 32 + lane) << 4]);
            oacc[nt]     = __builtin_amdgcn_wmma_f32_16x16x32_bf16(false, ay0, false, bf,
                                                                   (short)0, oacc[nt], false, false);
            oacc[8 + nt] = __builtin_amdgcn_wmma_f32_16x16x32_bf16(false, ay1, false, bf,
                                                                   (short)0, oacc[8 + nt], false, false);
        }
    }

    // ---- epilogue: fc2 bias + fp32 store (once per block, amortized over 128 chunks)
    {
        const int n  = lane & 15;
        const int rb = (lane >> 4) << 3;
        #pragma unroll
        for (int nt = 0; nt < 8; ++nt) {
            const int col = w * 128 + nt * 16 + n;
            const float bb = b2[(size_t)e * O_ + col];
            #pragma unroll
            for (int rt = 0; rt < 2; ++rt)
                #pragma unroll
                for (int r = 0; r < 8; ++r) {
                    const int row = rt * 16 + rb + r;
                    out[((size_t)e * C_ + row0 + row) * O_ + col] = oacc[rt * 8 + nt][r] + bb;
                }
        }
    }
}

extern "C" void kernel_launch(void* const* d_in, const int* in_sizes, int n_in,
                              void* d_out, int out_size, void* d_ws, size_t ws_size,
                              hipStream_t stream) {
    (void)in_sizes; (void)n_in; (void)out_size; (void)d_ws; (void)ws_size;
    const float* x  = (const float*)d_in[0];
    const float* w1 = (const float*)d_in[1];
    const float* b1 = (const float*)d_in[2];
    const float* w2 = (const float*)d_in[3];
    const float* b2 = (const float*)d_in[4];
    float* out = (float*)d_out;

    const size_t shmem = (size_t)(LDS_X_SHORTS + LDS_W1_SHORTS + LDS_W2_SHORTS
                                  + LDS_YF_SHORTS) * sizeof(unsigned short)
                       + (size_t)LDS_PART_FLTS * sizeof(float);     // ~226 KB

    dim3 grid(E_ * (C_ / BC));   // 8 experts * 128 slabs = 1024 blocks
    dim3 block(256);             // 8 wave32s
    hipLaunchKernelGGL(moe_ffn_fused_kernel, grid, block, shmem, stream,
                       x, w1, b1, w2, b2, out);
}
